// BiLstmCrf_28389733826574
// MI455X (gfx1250) — matrix operand order repaired and checked by reading, not compiled, not run on hardware
//
#include <hip/hip_runtime.h>
#include <hip/hip_bf16.h>
#include <stdint.h>

typedef unsigned short u16;
typedef unsigned int   u32;
typedef __attribute__((ext_vector_type(16))) __bf16 v16bf;
typedef __attribute__((ext_vector_type(8)))  float  v8f;

#define B_   64
#define T_   512
#define D_   300
#define DP_  320      // D padded to multiple of 32 for WMMA K
#define H_   128
#define G_   512      // 4*H gates
#define HD_  256      // 2*H bidirectional
#define L_   9
#define M_   (T_*B_)  // 32768 rows of x
#define BG_  16       // batch rows per recurrence workgroup

__device__ __forceinline__ u16 f2bf(float f){
  u32 u = __float_as_uint(f);
  u32 r = (u + 0x7FFFu + ((u >> 16) & 1u)) >> 16;   // round-to-nearest-even
  return (u16)r;
}
__device__ __forceinline__ float bf2f(u16 h){
  return __uint_as_float(((u32)h) << 16);
}
__device__ __forceinline__ float fast_rcp(float x){
  return __builtin_amdgcn_rcpf(x);
}
__device__ __forceinline__ float sigmoidf_(float x){
  return fast_rcp(1.0f + __expf(-x));
}
__device__ __forceinline__ float tanhf_(float x){
  float e = __expf(2.0f * x);
  return 1.0f - 2.0f * fast_rcp(e + 1.0f);
}

// ---------------- WMMA bf16 16x16x32 fragment loaders -----------------------
// A fragment: 16(M) x 32(K) from row-major src, leading dim ld (elements).
// Lanes 0-15: M=0-15 / K=0-15; lanes 16-31: M=0-15 / K=16-31.
// Elem e -> K = base+e  (32B contiguous -> 2x b128)
__device__ __forceinline__ v16bf load_frag_a(const u16* base, int ld, int lane){
  int m = lane & 15, hi = lane >> 4;
  const u16* p = base + (size_t)m * ld + hi * 16;
  union { v16bf v; u16 u[16]; } r;
#pragma unroll
  for (int e = 0; e < 16; ++e) r.u[e] = p[e];
  return r.v;
}
// B fragment: 32(K) x 16(N); source is weight matrix row-major [N][K] (= B^T),
// ld = K stride.  Lanes 0-15: N=lane, K=0-15; lanes 16-31: N=lane-16, K=16-31.
__device__ __forceinline__ v16bf load_frag_b(const u16* base, int ld, int lane){
  int n = lane & 15, hi = lane >> 4;
  const u16* p = base + (size_t)n * ld + hi * 16;
  union { v16bf v; u16 u[16]; } r;
#pragma unroll
  for (int e = 0; e < 16; ++e) r.u[e] = p[e];   // 32B contiguous -> 2x b128
  return r.v;
}

// ---------------- small prep kernels ----------------------------------------
__global__ void k_cvt_pad(const float* __restrict__ src, u16* __restrict__ dst,
                          int rows, int kin, int kout){
  int idx = blockIdx.x * blockDim.x + threadIdx.x;
  if (idx >= rows * kout) return;
  int k = idx % kout, r = idx / kout;
  float v = (k < kin) ? src[(size_t)r * kin + k] : 0.0f;
  dst[idx] = f2bf(v);
}

__global__ void k_bias_sum(const float* __restrict__ a, const float* __restrict__ b,
                           float* __restrict__ o, int n){
  int i = blockIdx.x * blockDim.x + threadIdx.x;
  if (i < n) o[i] = a[i] + b[i];
}

// x[t*B+b][k] = emb[word[b][t]][k], padded K, bf16
__global__ void k_gather_embed(const int* __restrict__ word,
                               const float* __restrict__ emb,
                               u16* __restrict__ x){
  long long idx = (long long)blockIdx.x * blockDim.x + threadIdx.x;
  if (idx >= (long long)M_ * DP_) return;
  int k   = (int)(idx % DP_);
  int row = (int)(idx / DP_);        // row = t*B + b
  int t = row / B_, b = row % B_;
  int w = word[b * T_ + t];
  float v = (k < D_) ? emb[(size_t)w * D_ + k] : 0.0f;
  x[idx] = f2bf(v);
}

// ---------------- input projection GEMM: xp = A @ W^T + bias ----------------
// A: bf16 [M_][Kpad] row-major.  W: bf16 [2][G_][Kpad] row-major.
// C: f32 [2][M_][H_][4]  (gate-interleaved: ((row*128 + hc)*4 + gate)).
// Each wave computes a 16x64 strip (4 C tiles) reusing one A fragment.
__global__ void __launch_bounds__(256)
k_gemm_inproj(const u16* __restrict__ A, const u16* __restrict__ W,
              const float* __restrict__ bias, float* __restrict__ C, int Kpad){
  int lane  = threadIdx.x & 31;
  int wave  = threadIdx.x >> 5;
  int mTile = blockIdx.x;
  int dir   = blockIdx.z;
  int nBase = wave * 64;                     // 4 nTiles of 16 per wave
  const u16*   Wd = W + (size_t)dir * G_ * Kpad;
  const float* bd = bias + dir * G_;
  float*       Cd = C + (size_t)dir * M_ * G_;
  const u16* Arow = A  + (size_t)mTile * 16 * Kpad;
  const u16* W0   = Wd + (size_t)nBase * Kpad;
  v8f acc[4] = {{}, {}, {}, {}};
  for (int k = 0; k < Kpad; k += 32){
    v16bf a = load_frag_a(Arow + k, Kpad, lane);
#pragma unroll
    for (int j = 0; j < 4; ++j){
      v16bf b = load_frag_b(W0 + (size_t)j * 16 * Kpad + k, Kpad, lane);
      acc[j] = __builtin_amdgcn_wmma_f32_16x16x32_bf16(false, a, false, b,
                                                       (short)0, acc[j], false, false);
    }
  }
  int hi = lane >> 4;
#pragma unroll
  for (int j = 0; j < 4; ++j){
    int n  = nBase + j * 16 + (lane & 15);
    int q  = n >> 7;            // gate index 0..3
    int hc = n & (H_ - 1);      // hidden column
    float bn = bd[n];
#pragma unroll
    for (int r = 0; r < 8; ++r){
      int m = mTile * 16 + r + hi * 8;
      Cd[((size_t)m * H_ + hc) * 4 + q] = acc[j][r] + bn;
    }
  }
}

// ---------------- persistent batch-split LSTM recurrence --------------------
// grid = (4 batch-groups, 2 dirs), block = 256 threads = 8 waves.
// Each WG owns 16 batch rows; each wave owns one 16-wide h column tile and
// computes its i/f/g/o gate tiles.  W_hh fragments are hoisted into registers
// (time-invariant); h lives in 4KB LDS; c is register-persistent.
__global__ void __launch_bounds__(256)
k_lstm_dir(const float* __restrict__ xp, const u16* __restrict__ Whh,
           u16* __restrict__ hout){
  __shared__ u16 hS[BG_ * H_];               // 4 KB
  const int bg    = blockIdx.x;              // batch group 0..3
  const int dir   = blockIdx.y;
  const int lane  = threadIdx.x & 31;
  const int hTile = threadIdx.x >> 5;        // wave id = h column tile 0..7
  const float* xpd = xp  + (size_t)dir * M_ * G_;
  const u16*   Wd  = Whh + (size_t)dir * G_ * H_;
  const int colBase = dir * H_;
  const int b0 = bg * BG_;

  for (int i = threadIdx.x; i < BG_ * H_; i += blockDim.x) hS[i] = 0;

  // hoist the 16 time-invariant W_hh B-fragments (4 gates x 4 K-steps)
  v16bf Bf[4][4];
#pragma unroll
  for (int q = 0; q < 4; ++q)
#pragma unroll
    for (int kk = 0; kk < 4; ++kk)
      Bf[q][kk] = load_frag_b(Wd + (size_t)(q * H_ + hTile * 16) * H_ + kk * 32,
                              H_, lane);
  __syncthreads();

  v8f c8 = {};
  const int hc  = hTile * 16 + (lane & 15);
  const int hi2 = lane >> 4;

  for (int ti = 0; ti < T_; ++ti){
    int t = (dir == 0) ? ti : (T_ - 1 - ti);
    // prefetch next step's 32KB xp strip for this batch group
    if (ti + 1 < T_){
      int tn = (dir == 0) ? (ti + 1) : (T_ - 2 - ti);
      const char* pf = (const char*)(xpd + ((size_t)tn * B_ + b0) * G_);
      __builtin_prefetch(pf + (size_t)threadIdx.x * 128, 0, 1);
    }
    v8f gi = {}, gf = {}, gg = {}, go = {};
#pragma unroll
    for (int kk = 0; kk < 4; ++kk){
      v16bf a = load_frag_a(&hS[kk * 32], H_, lane);
      gi = __builtin_amdgcn_wmma_f32_16x16x32_bf16(false, a, false, Bf[0][kk], (short)0, gi, false, false);
      gf = __builtin_amdgcn_wmma_f32_16x16x32_bf16(false, a, false, Bf[1][kk], (short)0, gf, false, false);
      gg = __builtin_amdgcn_wmma_f32_16x16x32_bf16(false, a, false, Bf[2][kk], (short)0, gg, false, false);
      go = __builtin_amdgcn_wmma_f32_16x16x32_bf16(false, a, false, Bf[3][kk], (short)0, go, false, false);
    }
    v8f h8 = {};
#pragma unroll
    for (int r = 0; r < 8; ++r){
      int b = b0 + r + hi2 * 8;
      const float4 g4 = *(const float4*)(xpd + (((size_t)t * B_ + b) * H_ + hc) * 4);
      float vi = sigmoidf_(gi[r] + g4.x);
      float vf = sigmoidf_(gf[r] + g4.y);
      float vg = tanhf_   (gg[r] + g4.z);
      float vo = sigmoidf_(go[r] + g4.w);
      float cc = vf * c8[r] + vi * vg;
      c8[r] = cc;
      h8[r] = vo * tanhf_(cc);
    }
    __syncthreads();                 // all waves done reading old h
#pragma unroll
    for (int r = 0; r < 8; ++r){
      int bl = r + hi2 * 8;          // local batch row 0..15
      u16 hb = f2bf(h8[r]);
      hS[bl * H_ + hc] = hb;
      hout[((size_t)t * B_ + b0 + bl) * HD_ + colBase + hc] = hb;
    }
    __syncthreads();                 // new h visible for next step
  }
}

// ---------------- emission projection: em[b][t][l] --------------------------
__global__ void k_emproj(const u16* __restrict__ h2, const float* __restrict__ wOut,
                         const float* __restrict__ bOut, float* __restrict__ em){
  int idx = blockIdx.x * blockDim.x + threadIdx.x;
  if (idx >= B_ * T_ * L_) return;
  int l  = idx % L_;
  int bt = idx / L_;
  int t  = bt % T_;
  int b  = bt / T_;
  const u16*   x = h2 + ((size_t)t * B_ + b) * HD_;
  const float* w = wOut + l * HD_;
  float s = bOut[l];
#pragma unroll 8
  for (int k = 0; k < HD_; ++k) s += bf2f(x[k]) * w[k];
  em[idx] = s;          // layout (b*T + t)*L + l
}

// ---------------- CRF negative log-likelihood -------------------------------
__global__ void __launch_bounds__(64)
k_crf(const int* __restrict__ word, const int* __restrict__ tags,
      const float* __restrict__ em, const float* __restrict__ start_t,
      const float* __restrict__ end_t, const float* __restrict__ trans,
      float* __restrict__ out){
  __shared__ float trS[L_*L_], stS[L_], enS[L_], partial[B_];
  int tid = threadIdx.x;
  if (tid < L_*L_) trS[tid] = trans[tid];
  if (tid < L_) { stS[tid] = start_t[tid]; enS[tid] = end_t[tid]; }
  __syncthreads();

  int b = tid;
  const int*   wrow = word + b * T_;
  const int*   trow = tags + b * T_;
  const float* erow = em + (size_t)b * T_ * L_;

  // numerator
  int tag0 = trow[0];
  float num = stS[tag0] + erow[tag0];
  int prev = tag0;
  int cnt = (wrow[0] > 0) ? 1 : 0;
  for (int t = 1; t < T_; ++t){
    int tg = trow[t];
    float m = (wrow[t] > 0) ? 1.0f : 0.0f;
    num += (trS[prev * L_ + tg] + erow[t * L_ + tg]) * m;
    prev = tg;
    cnt += (wrow[t] > 0) ? 1 : 0;
  }
  int lastIdx = cnt - 1; if (lastIdx < 0) lastIdx = 0;
  num += enS[trow[lastIdx]];

  // forward algorithm (partition function)
  float alpha[L_];
#pragma unroll
  for (int j = 0; j < L_; ++j) alpha[j] = stS[j] + erow[j];
  for (int t = 1; t < T_; ++t){
    if (wrow[t] > 0){
      float na[L_];
#pragma unroll
      for (int j = 0; j < L_; ++j){
        float mx = -1e30f;
#pragma unroll
        for (int i = 0; i < L_; ++i){
          float v = alpha[i] + trS[i * L_ + j];
          mx = fmaxf(mx, v);
        }
        float s = 0.0f;
#pragma unroll
        for (int i = 0; i < L_; ++i)
          s += __expf(alpha[i] + trS[i * L_ + j] - mx);
        na[j] = mx + __logf(s) + erow[t * L_ + j];
      }
#pragma unroll
      for (int j = 0; j < L_; ++j) alpha[j] = na[j];
    }
  }
  float mx = -1e30f;
#pragma unroll
  for (int j = 0; j < L_; ++j) mx = fmaxf(mx, alpha[j] + enS[j]);
  float s = 0.0f;
#pragma unroll
  for (int j = 0; j < L_; ++j) s += __expf(alpha[j] + enS[j] - mx);
  float denom = mx + __logf(s);

  partial[b] = num - denom;
  __syncthreads();
  if (tid == 0){
    float acc = 0.0f;
    for (int i = 0; i < B_; ++i) acc += partial[i];
    out[0] = -acc;
  }
}

// ---------------- host-side launch ------------------------------------------
extern "C" void kernel_launch(void* const* d_in, const int* in_sizes, int n_in,
                              void* d_out, int out_size, void* d_ws, size_t ws_size,
                              hipStream_t stream){
  (void)in_sizes; (void)n_in; (void)out_size; (void)ws_size;
  const int*   word   = (const int*)d_in[0];
  const int*   label  = (const int*)d_in[1];
  const float* emb    = (const float*)d_in[2];
  const float* w_ih0  = (const float*)d_in[3];
  const float* w_hh0  = (const float*)d_in[4];
  const float* b_ih0  = (const float*)d_in[5];
  const float* b_hh0  = (const float*)d_in[6];
  const float* w_ih1  = (const float*)d_in[7];
  const float* w_hh1  = (const float*)d_in[8];
  const float* b_ih1  = (const float*)d_in[9];
  const float* b_hh1  = (const float*)d_in[10];
  const float* w_out  = (const float*)d_in[11];
  const float* b_out  = (const float*)d_in[12];
  const float* startT = (const float*)d_in[13];
  const float* endT   = (const float*)d_in[14];
  const float* trans  = (const float*)d_in[15];

  char* ws = (char*)d_ws;
  size_t off = 0;
  auto alloc = [&](size_t bytes) -> void* {
    void* p = ws + off;
    off = (off + bytes + 255) & ~(size_t)255;
    return p;
  };
  u16*   xbf   = (u16*)  alloc((size_t)M_ * DP_ * 2);      // ~21 MB
  u16*   w0b   = (u16*)  alloc((size_t)2 * G_ * DP_ * 2);
  u16*   wh0b  = (u16*)  alloc((size_t)2 * G_ * H_  * 2);
  u16*   w1b   = (u16*)  alloc((size_t)2 * G_ * HD_ * 2);
  u16*   wh1b  = (u16*)  alloc((size_t)2 * G_ * H_  * 2);
  float* bias0 = (float*)alloc((size_t)2 * G_ * 4);
  float* bias1 = (float*)alloc((size_t)2 * G_ * 4);
  float* xp    = (float*)alloc((size_t)2 * M_ * G_ * 4);   // 134 MB, reused by layer1
  u16*   h1    = (u16*)  alloc((size_t)M_ * HD_ * 2);      // ~17 MB
  u16*   h2    = (u16*)  alloc((size_t)M_ * HD_ * 2);      // ~17 MB
  float* em    = (float*)alloc((size_t)B_ * T_ * L_ * 4);  // ~1.2 MB

  const int blk = 256;
  // weight / bias prep
  k_cvt_pad<<<(2*G_*DP_ + blk-1)/blk, blk, 0, stream>>>(w_ih0, w0b,  2*G_, D_,  DP_);
  k_cvt_pad<<<(2*G_*H_  + blk-1)/blk, blk, 0, stream>>>(w_hh0, wh0b, 2*G_, H_,  H_);
  k_cvt_pad<<<(2*G_*HD_ + blk-1)/blk, blk, 0, stream>>>(w_ih1, w1b,  2*G_, HD_, HD_);
  k_cvt_pad<<<(2*G_*H_  + blk-1)/blk, blk, 0, stream>>>(w_hh1, wh1b, 2*G_, H_,  H_);
  k_bias_sum<<<(2*G_ + blk-1)/blk, blk, 0, stream>>>(b_ih0, b_hh0, bias0, 2*G_);
  k_bias_sum<<<(2*G_ + blk-1)/blk, blk, 0, stream>>>(b_ih1, b_hh1, bias1, 2*G_);
  // embedding gather (bf16, K padded)
  long long ng = (long long)M_ * DP_;
  k_gather_embed<<<(int)((ng + blk - 1) / blk), blk, 0, stream>>>(word, emb, xbf);
  // layer 0
  k_gemm_inproj<<<dim3(M_/16, 1, 2), 256, 0, stream>>>(xbf, w0b, bias0, xp, DP_);
  k_lstm_dir  <<<dim3(B_/BG_, 2), 256, 0, stream>>>(xp, wh0b, h1);
  // layer 1 (reuse xp buffer)
  k_gemm_inproj<<<dim3(M_/16, 1, 2), 256, 0, stream>>>(h1, w1b, bias1, xp, HD_);
  k_lstm_dir  <<<dim3(B_/BG_, 2), 256, 0, stream>>>(xp, wh1b, h2);
  // emissions + CRF
  int nem = B_ * T_ * L_;
  k_emproj<<<(nem + blk - 1)/blk, blk, 0, stream>>>(h2, w_out, b_out, em);
  k_crf<<<1, 64, 0, stream>>>(word, label, em, startT, endT, trans, (float*)d_out);
}
